// VisualAttentionHead_4303557230859
// MI455X (gfx1250) — matrix operand order repaired and checked by reading, not compiled, not run on hardware
//
#include <hip/hip_runtime.h>

#define BATCH 8
#define CIN   256
#define CQ    32      // q/k channels
#define CVCH  128     // v channels
#define NPIX  4096    // 64*64

typedef __attribute__((ext_vector_type(16))) __bf16 v16bf;
typedef __attribute__((ext_vector_type(8)))  float  v8f;

union V16 { v16bf v; unsigned int u[8]; };
union V8F { v8f   v; float        f[8]; };

// ---- bf16 conversion: single v_perm_b32 packs two f32 -> packed bf16 -------
static __device__ __forceinline__ unsigned int pack_bf16(float a, float b) {
  // result = { trunc_bf16(b), trunc_bf16(a) } : bytes {b.hi16, a.hi16}
  return __builtin_amdgcn_perm(__float_as_uint(b), __float_as_uint(a), 0x07060302u);
}
static __device__ __forceinline__ unsigned short f32_to_bf16(float f) {
  return (unsigned short)(__float_as_uint(f) >> 16);   // folds into *_d16_hi stores
}
static __device__ __forceinline__ v8f wmma_bf16(v16bf a, v16bf b, v8f c) {
  return __builtin_amdgcn_wmma_f32_16x16x32_bf16(false, a, false, b, (short)0, c, false, false);
}
// Cross-lane traffic through LDS inside one wave: force ds completion + ordering.
static __device__ __forceinline__ void lds_wave_fence() {
  asm volatile("s_wait_dscnt 0x0" ::: "memory");
}

// ---- async global->LDS (ASYNCcnt) with sync fallback -----------------------
#if __has_builtin(__builtin_amdgcn_global_load_async_to_lds_b128)
#define AS1 __attribute__((address_space(1)))
#define AS3 __attribute__((address_space(3)))
typedef int v4i_vs __attribute__((vector_size(16)));   // matches builtin param type
static __device__ __forceinline__ void copy16_async(const void* g, void* l) {
  __builtin_amdgcn_global_load_async_to_lds_b128((AS1 v4i_vs*)g, (AS3 v4i_vs*)l, 0, 0);
}
static __device__ __forceinline__ void wait_async() {
#if __has_builtin(__builtin_amdgcn_s_wait_asynccnt)
  __builtin_amdgcn_s_wait_asynccnt(0);
#else
  asm volatile("s_wait_asynccnt 0x0" ::: "memory");
#endif
}
#else
static __device__ __forceinline__ void copy16_async(const void* g, void* l) {
  *(uint4*)l = *(const uint4*)g;
}
static __device__ __forceinline__ void wait_async() {}
#endif

// ---------------------------------------------------------------------------
// Kernel 1: per-pixel QKV projection (GEMM [64 pix x 256] x [256 x 192]).
// 12 waves/block; wave w owns output tile w (0-1: q, 2-3: k, 4-11: v),
// keeps its full-K B-matrix persistent in registers.
// ---------------------------------------------------------------------------
__global__ __launch_bounds__(384)
void vah_qkv_proj(const float* __restrict__ x,
                  const float* __restrict__ Wq, const float* __restrict__ bq,
                  const float* __restrict__ Wk, const float* __restrict__ bk,
                  const float* __restrict__ Wv, const float* __restrict__ bv,
                  unsigned short* __restrict__ qbf,
                  unsigned short* __restrict__ kbf,
                  unsigned short* __restrict__ vT,
                  float* __restrict__ out) {
  __shared__ alignas(16) unsigned short lds_x[64 * 258];  // [pixel][ch], pad 2

  const int b    = blockIdx.y;
  const int n0   = blockIdx.x * 64;
  const int tid  = threadIdx.x;
  const int wave = tid >> 5;
  const int lane = tid & 31;
  const int col  = lane & 15;
  const int half = lane >> 4;

  // Stage x tile: xf[p][ch] = x[b][ch][n0+p]; convert channel pairs -> one b32 store.
  for (int idx = tid; idx < 64 * 128; idx += 384) {
    int p = idx & 63, c2 = idx >> 6;
    float a0 = x[((size_t)b * CIN + 2 * c2 + 0) * NPIX + n0 + p];
    float a1 = x[((size_t)b * CIN + 2 * c2 + 1) * NPIX + n0 + p];
    *(unsigned int*)&lds_x[p * 258 + 2 * c2] = pack_bf16(a0, a1);
  }
  __syncthreads();

  const float* Wsrc; const float* bsrc; int obase;
  if (wave < 2)      { Wsrc = Wq; bsrc = bq; obase = wave * 16; }
  else if (wave < 4) { Wsrc = Wk; bsrc = bk; obase = (wave - 2) * 16; }
  else               { Wsrc = Wv; bsrc = bv; obase = (wave - 4) * 16; }
  // kind is wave-uniform: force scalar branches instead of exec masking.
  const int kind = __builtin_amdgcn_readfirstlane(wave < 2 ? 0 : (wave < 4 ? 1 : 2));

  const int o = obase + col;           // this lane's output channel (B col)
  // Persistent B: 8 chunks of K=32.  B layout: col = l%16, K pairs 2*(v&3)+16*(v>>2)+8*half.
  V16 Bw[8];
  #pragma unroll
  for (int c = 0; c < 8; ++c)
    #pragma unroll
    for (int v = 0; v < 8; ++v) {
      const float* wp = Wsrc + (size_t)o * CIN + c * 32 + 2 * (v & 3) + ((v & 4) ? 16 : 0) + (half ? 8 : 0);
      Bw[c].u[v] = pack_bf16(wp[0], wp[1]);
    }
  const float bias = bsrc[o];

  for (int pt = 0; pt < 4; ++pt) {
    const int p = pt * 16 + col;       // A row M = l%16
    // Hoist all A chunks so DS loads pipeline ahead of the WMMA chain.
    V16 A[8];
    #pragma unroll
    for (int c = 0; c < 8; ++c)
      #pragma unroll
      for (int v = 0; v < 8; ++v) {
        int ch = c * 32 + 2 * (v & 3) + ((v & 4) ? 16 : 0) + (half ? 8 : 0);
        A[c].u[v] = *(const unsigned int*)&lds_x[p * 258 + ch];
      }
    V8F acc;
    #pragma unroll
    for (int i = 0; i < 8; ++i) acc.f[i] = 0.0f;
    #pragma unroll
    for (int c = 0; c < 8; ++c) acc.v = wmma_bf16(A[c].v, Bw[c].v, acc.v);

    if (kind == 0) {
      #pragma unroll
      for (int v = 0; v < 8; ++v) {
        int pix = n0 + pt * 16 + v + 8 * half;
        qbf[((size_t)b * NPIX + pix) * CQ + o] = f32_to_bf16(acc.f[v] + bias);
      }
    } else if (kind == 1) {
      #pragma unroll
      for (int v = 0; v < 8; ++v) {
        int pix = n0 + pt * 16 + v + 8 * half;
        kbf[((size_t)b * NPIX + pix) * CQ + o] = f32_to_bf16(acc.f[v] + bias);
      }
    } else {
      #pragma unroll
      for (int v = 0; v < 8; ++v) {
        int pix = n0 + pt * 16 + v + 8 * half;
        float r = fmaxf(acc.f[v] + bias, 0.0f);                  // relu(v)
        vT[((size_t)b * CVCH + o) * NPIX + pix] = f32_to_bf16(r);
        out[((size_t)b * CIN + CVCH + o) * NPIX + pix] = r;      // concat second half
      }
    }
  }
}

// ---------------------------------------------------------------------------
// Kernel 2: flash attention + output projection.
// 8 waves/block, 16 queries/wave. Double-buffered async K/V staging in LDS.
// ---------------------------------------------------------------------------
__global__ __launch_bounds__(256)
void vah_flash_attn(const unsigned short* __restrict__ qbf,
                    const unsigned short* __restrict__ kbf,
                    const unsigned short* __restrict__ vT,
                    const float* __restrict__ Wc, const float* __restrict__ bc,
                    float* __restrict__ out) {
  // Loop-time buffers overlap the epilogue O-transpose scratch (union).
  __shared__ alignas(16) union {
    struct {
      unsigned short k[2][32 * 40];    // [key][ch], stride 40 -> conflict-free
      unsigned short v[2][128 * 40];   // [d][key]
      unsigned short p[8][16 * 40];    // per-wave P transpose scratch
    } loop;
    unsigned short o[8][16 * 136];     // per-wave O transpose scratch
  } lds;

  const int b    = blockIdx.y;
  const int q0   = blockIdx.x * 128;
  const int tid  = threadIdx.x;
  const int wave = tid >> 5;
  const int lane = tid & 31;
  const int col  = lane & 15;
  const int half = lane >> 4;
  const int qrow = q0 + wave * 16 + col;

  // Q as A-matrix (loaded once): contiguous bf16 pairs.
  V16 Aq;
  #pragma unroll
  for (int v = 0; v < 8; ++v) {
    int ch = 2 * (v & 3) + ((v & 4) ? 16 : 0) + (half ? 8 : 0);
    Aq.u[v] = *(const unsigned int*)&qbf[((size_t)b * NPIX + qrow) * CQ + ch];
  }

  // Staging assignment (constant per thread).
  const int skey = tid >> 2, sseg = (tid & 3) * 8;     // K tile, threads 0..127
  const int sd   = tid >> 1, svseg = (tid & 1) * 16;   // V tile, all threads

  float m_run[8], l_run[8];
  V8F O[8];
  #pragma unroll
  for (int v = 0; v < 8; ++v) { m_run[v] = -1e30f; l_run[v] = 0.0f; }
  #pragma unroll
  for (int dt = 0; dt < 8; ++dt)
    #pragma unroll
    for (int v = 0; v < 8; ++v) O[dt].f[v] = 0.0f;

  // Prologue: async-stage tile kk=0 into buffer 0.
  if (tid < 128)
    copy16_async(&kbf[((size_t)b * NPIX + 0 + skey) * CQ + sseg],
                 &lds.loop.k[0][skey * 40 + sseg]);
  {
    const unsigned short* src = &vT[((size_t)b * CVCH + sd) * NPIX + 0 + svseg];
    copy16_async(src,     &lds.loop.v[0][sd * 40 + svseg]);
    copy16_async(src + 8, &lds.loop.v[0][sd * 40 + svseg + 8]);
  }

  for (int kk = 0; kk < NPIX; kk += 32) {
    const int buf = (kk >> 5) & 1;
    wait_async();                       // our async writes into buf are done
    __syncthreads();                    // whole block: buf ready, prev reads done
    if (kk + 32 < NPIX) {               // overlap next tile's staging with compute
      const int nb = buf ^ 1, nk = kk + 32;
      if (tid < 128)
        copy16_async(&kbf[((size_t)b * NPIX + nk + skey) * CQ + sseg],
                     &lds.loop.k[nb][skey * 40 + sseg]);
      const unsigned short* src = &vT[((size_t)b * CVCH + sd) * NPIX + nk + svseg];
      copy16_async(src,     &lds.loop.v[nb][sd * 40 + svseg]);
      copy16_async(src + 8, &lds.loop.v[nb][sd * 40 + svseg + 8]);
    }
    const unsigned short* ck = lds.loop.k[buf];
    const unsigned short* cv = lds.loop.v[buf];

    // S = Q K^T : two 16-key sub-tiles
    V16 Bk0, Bk1;
    #pragma unroll
    for (int v = 0; v < 8; ++v) {
      int ch = 2 * (v & 3) + ((v & 4) ? 16 : 0) + (half ? 8 : 0);
      Bk0.u[v] = *(const unsigned int*)&ck[col * 40 + ch];
      Bk1.u[v] = *(const unsigned int*)&ck[(col + 16) * 40 + ch];
    }
    v8f z = {};
    V8F s0, s1;
    s0.v = wmma_bf16(Aq.v, Bk0.v, z);
    s1.v = wmma_bf16(Aq.v, Bk1.v, z);

    // Online softmax; C rows m = v + 8*half, cols = keys across 16 lanes.
    float mt[8];
    #pragma unroll
    for (int v = 0; v < 8; ++v) mt[v] = fmaxf(s0.f[v], s1.f[v]);
    #pragma unroll
    for (int off = 1; off < 16; off <<= 1)
      #pragma unroll
      for (int v = 0; v < 8; ++v) mt[v] = fmaxf(mt[v], __shfl_xor(mt[v], off, 32));

    float alpha[8], sum[8];
    #pragma unroll
    for (int v = 0; v < 8; ++v) {
      float mn = fmaxf(m_run[v], mt[v]);
      alpha[v] = __expf(m_run[v] - mn);
      m_run[v] = mn;
      s0.f[v] = __expf(s0.f[v] - mn);
      s1.f[v] = __expf(s1.f[v] - mn);
      sum[v]  = s0.f[v] + s1.f[v];
    }
    #pragma unroll
    for (int off = 1; off < 16; off <<= 1)
      #pragma unroll
      for (int v = 0; v < 8; ++v) sum[v] += __shfl_xor(sum[v], off, 32);
    #pragma unroll
    for (int v = 0; v < 8; ++v) l_run[v] = l_run[v] * alpha[v] + sum[v];
    #pragma unroll
    for (int dt = 0; dt < 8; ++dt)
      #pragma unroll
      for (int v = 0; v < 8; ++v) O[dt].f[v] *= alpha[v];

    // P: C-layout -> A-layout via per-wave LDS scratch.
    unsigned short* pbuf = lds.loop.p[wave];
    #pragma unroll
    for (int v = 0; v < 8; ++v) {
      int qr = v + 8 * half;
      pbuf[qr * 40 + col]      = f32_to_bf16(s0.f[v]);
      pbuf[qr * 40 + 16 + col] = f32_to_bf16(s1.f[v]);
    }
    lds_wave_fence();
    V16 Ap;
    #pragma unroll
    for (int v = 0; v < 8; ++v) {
      int kc = 2 * (v & 3) + ((v & 4) ? 16 : 0) + (half ? 8 : 0);
      Ap.u[v] = *(const unsigned int*)&pbuf[col * 40 + kc];
    }

    // O += P V : preload all 8 B tiles, then back-to-back WMMAs.
    V16 Bv[8];
    #pragma unroll
    for (int dt = 0; dt < 8; ++dt)
      #pragma unroll
      for (int v = 0; v < 8; ++v) {
        int kr = 2 * (v & 3) + ((v & 4) ? 16 : 0) + (half ? 8 : 0);
        Bv[dt].u[v] = *(const unsigned int*)&cv[(dt * 16 + col) * 40 + kr];
      }
    #pragma unroll
    for (int dt = 0; dt < 8; ++dt) O[dt].v = wmma_bf16(Ap.v, Bv[dt].v, O[dt].v);
  }

  __syncthreads();   // loop buffers die; union o-region becomes live

  // Normalize, transpose O to A-layout, comb = O Wc^T + bc.
  #pragma unroll
  for (int v = 0; v < 8; ++v) l_run[v] = 1.0f / l_run[v];
  unsigned short* obuf = lds.o[wave];
  #pragma unroll
  for (int dt = 0; dt < 8; ++dt)
    #pragma unroll
    for (int v = 0; v < 8; ++v) {
      int qr = v + 8 * half;
      obuf[qr * 136 + dt * 16 + col] = f32_to_bf16(O[dt].f[v] * l_run[v]);
    }
  lds_wave_fence();

  V16 Ao[4];
  #pragma unroll
  for (int c = 0; c < 4; ++c)
    #pragma unroll
    for (int v = 0; v < 8; ++v) {
      int d = c * 32 + 2 * (v & 3) + ((v & 4) ? 16 : 0) + (half ? 8 : 0);
      Ao[c].u[v] = *(const unsigned int*)&obuf[col * 136 + d];
    }

  #pragma unroll
  for (int ot = 0; ot < 8; ++ot) {
    const int o = ot * 16 + col;
    V16 Bw[4];
    #pragma unroll
    for (int c = 0; c < 4; ++c)
      #pragma unroll
      for (int v = 0; v < 8; ++v) {
        const float* wp = Wc + (size_t)o * CVCH + c * 32 + 2 * (v & 3) + ((v & 4) ? 16 : 0) + (half ? 8 : 0);
        Bw[c].u[v] = pack_bf16(wp[0], wp[1]);
      }
    V8F acc;
    #pragma unroll
    for (int i = 0; i < 8; ++i) acc.f[i] = 0.0f;
    #pragma unroll
    for (int c = 0; c < 4; ++c) acc.v = wmma_bf16(Ao[c].v, Bw[c].v, acc.v);
    const float bias = bc[o];
    #pragma unroll
    for (int v = 0; v < 8; ++v) {
      int q = q0 + wave * 16 + v + 8 * half;
      out[((size_t)b * CIN + o) * NPIX + q] = acc.f[v] + bias;
    }
  }
}

// ---------------------------------------------------------------------------
extern "C" void kernel_launch(void* const* d_in, const int* in_sizes, int n_in,
                              void* d_out, int out_size, void* d_ws, size_t ws_size,
                              hipStream_t stream) {
  (void)in_sizes; (void)n_in; (void)out_size; (void)ws_size;
  const float* x  = (const float*)d_in[0];
  const float* Wq = (const float*)d_in[1];
  const float* bq = (const float*)d_in[2];
  const float* Wk = (const float*)d_in[3];
  const float* bk = (const float*)d_in[4];
  const float* Wv = (const float*)d_in[5];
  const float* bv = (const float*)d_in[6];
  const float* Wc = (const float*)d_in[7];
  const float* bc = (const float*)d_in[8];
  float* out = (float*)d_out;

  unsigned short* qbf = (unsigned short*)d_ws;                       // 2 MB
  unsigned short* kbf = qbf + (size_t)BATCH * NPIX * CQ;             // 2 MB
  unsigned short* vT  = kbf + (size_t)BATCH * NPIX * CQ;             // 8 MB

  dim3 g1(NPIX / 64, BATCH);
  vah_qkv_proj<<<g1, 384, 0, stream>>>(x, Wq, bq, Wk, bk, Wv, bv, qbf, kbf, vT, out);

  dim3 g2(NPIX / 128, BATCH);
  vah_flash_attn<<<g2, 256, 0, stream>>>(qbf, kbf, vT, Wc, bc, out);
}
